// FastSelfAttention_5763846111939
// MI455X (gfx1250) — matrix operand order repaired and checked by reading, not compiled, not run on hardware
//
#include <hip/hip_runtime.h>
#include <hip/hip_bf16.h>

// ---------------------------------------------------------------------------
// FastSelfAttention for MI455X (gfx1250, wave32, WMMA)
//   B=4, S=4096, D=1024
//   GEMMs via v_wmma_f32_16x16x32_bf16 with ping-pong register double
//   buffering (fully unrolled k-loop -> no inter-buffer copies, immediate
//   load offsets); pooling scans parallelized over B*D channels;
//   denominators stored as reciprocals.
// ---------------------------------------------------------------------------

typedef __attribute__((ext_vector_type(16))) __bf16 v16bf;
typedef __attribute__((ext_vector_type(8)))  __bf16 v8bf;
typedef __attribute__((ext_vector_type(8)))  float  v8f;

#define BATCH 4
#define SEQ   4096
#define DIM   1024
#define ROWS  (BATCH * SEQ)          // 16384
#define KSTEPS (DIM / 32)            // 32
#define INV_SQRT_D 0.03125f          // 1/sqrt(1024)

// ---------------- block reduction (256 threads = 8 waves, wave32) ----------
__device__ __forceinline__ float block_reduce_sum(float v, float* smem) {
    const int lane = threadIdx.x & 31;
    const int wid  = threadIdx.x >> 5;
    #pragma unroll
    for (int off = 16; off > 0; off >>= 1) v += __shfl_down(v, off, 32);
    if (lane == 0) smem[wid] = v;
    __syncthreads();
    if (wid == 0) {
        float t = (lane < (int)(blockDim.x >> 5)) ? smem[lane] : 0.0f;
        #pragma unroll
        for (int off = 16; off > 0; off >>= 1) t += __shfl_down(t, off, 32);
        if (lane == 0) smem[0] = t;
    }
    __syncthreads();
    float r = smem[0];
    __syncthreads();           // safe for back-to-back calls
    return r;
}

// ---------------- 1) LayerNorm + cast to bf16 -------------------------------
__global__ __launch_bounds__(256)
void ln_cast_kernel(const float* __restrict__ x, const float* __restrict__ g,
                    const float* __restrict__ b, __bf16* __restrict__ h) {
    __shared__ float smem[8];
    const size_t row = blockIdx.x;
    const float* xr = x + row * (size_t)DIM;
    float4 v = *(const float4*)(xr + threadIdx.x * 4);
    float s  = v.x + v.y + v.z + v.w;
    float sq = v.x*v.x + v.y*v.y + v.z*v.z + v.w*v.w;
    float tot   = block_reduce_sum(s,  smem);
    float totsq = block_reduce_sum(sq, smem);
    float mu  = tot * (1.0f / DIM);
    float var = totsq * (1.0f / DIM) - mu * mu;
    float rs  = rsqrtf(var + 1e-5f);
    const int d0 = threadIdx.x * 4;
    __bf16* hr = h + row * (size_t)DIM;
    const float xv[4] = {v.x, v.y, v.z, v.w};
    #pragma unroll
    for (int i = 0; i < 4; ++i) {
        float y = (xv[i] - mu) * rs * g[d0 + i] + b[d0 + i];
        hr[d0 + i] = (__bf16)y;
    }
}

// ---------------- 2) cast weights to bf16 -----------------------------------
__global__ __launch_bounds__(256)
void castw_kernel(const float* __restrict__ Wq, const float* __restrict__ Wk,
                  __bf16* __restrict__ wqb, __bf16* __restrict__ wkb, int n) {
    int i = blockIdx.x * blockDim.x + threadIdx.x;
    if (i < n) { wqb[i] = (__bf16)Wq[i]; wkb[i] = (__bf16)Wk[i]; }
}

// ---------------- 3) WMMA GEMM: C[m,n] = sum_k A[m,k] * W[n,k] --------------
// Block: 256 threads = 8 waves. Each wave -> 16(M) x 64(N) tile (4 accums).
// Block tile: 128(M) x 64(N). grid.z picks {Wq->Qout, Wk->Kout}.
// Fully unrolled k-loop with ping-pong fragment buffers: fragments for step
// it+1 are in flight while the four WMMAs for step it execute; buffer index
// (it & 1) is a compile-time constant so no register copies are generated and
// all global loads use immediate offsets.
__global__ __launch_bounds__(256)
void gemm_wmma_kernel(const __bf16* __restrict__ A,
                      const __bf16* __restrict__ Wqb,
                      const __bf16* __restrict__ Wkb,
                      float* __restrict__ Qout, float* __restrict__ Kout) {
    const __bf16* Bm = (blockIdx.z == 0) ? Wqb : Wkb;
    float*        C  = (blockIdx.z == 0) ? Qout : Kout;

    const int wave = threadIdx.x >> 5;      // 0..7
    const int lane = threadIdx.x & 31;
    const int half = lane >> 4;             // 0/1
    const int l16  = lane & 15;

    const int m0 = blockIdx.x * 128 + wave * 16;
    const int n0 = blockIdx.y * 64;

    // A fragment row for this lane (contiguous k order, same as B):
    //   elems 0..7  -> K = k0 + half*16 + i
    //   elems 8..15 -> K = k0 + half*16 + 8 + (i-8)
    const __bf16* arow = A + (size_t)(m0 + l16) * DIM + half * 16;
    // B fragment rows (32x16, K-major per lane): lane n = n0+16j+l16 holds
    // 16 contiguous K values starting at k0 + half*16 (== W[n, k..k+15]).
    const __bf16* brow[4];
    #pragma unroll
    for (int j = 0; j < 4; ++j)
        brow[j] = Bm + (size_t)(n0 + 16 * j + l16) * DIM + half * 16;

    auto load_a = [&](int k0) -> v16bf {
        union { v16bf v; v8bf h[2]; } au;
        au.h[0] = *(const v8bf*)(arow + k0);
        au.h[1] = *(const v8bf*)(arow + k0 + 8);
        return au.v;
    };

    v8f acc[4];
    #pragma unroll
    for (int j = 0; j < 4; ++j) acc[j] = 0.0f;

    v16bf a_frag[2];
    v16bf b_frag[2][4];

    // prologue: load k-step 0 into buffer 0
    a_frag[0] = load_a(0);
    #pragma unroll
    for (int j = 0; j < 4; ++j) b_frag[0][j] = *(const v16bf*)(brow[j]);

    #pragma unroll
    for (int it = 0; it < KSTEPS; ++it) {
        const int cur = it & 1;
        const int nxt = cur ^ 1;
        if (it + 1 < KSTEPS) {
            const int kn = (it + 1) * 32;
            a_frag[nxt] = load_a(kn);
            #pragma unroll
            for (int j = 0; j < 4; ++j)
                b_frag[nxt][j] = *(const v16bf*)(brow[j] + kn);
        }
        #pragma unroll
        for (int j = 0; j < 4; ++j)
            acc[j] = __builtin_amdgcn_wmma_f32_16x16x32_bf16(
                false, a_frag[cur], false, b_frag[cur][j], (short)0, acc[j],
                false, false);
    }

    // C/D layout: lane holds col n = l16; VGPR v -> row m0 + v + 8*half
    #pragma unroll
    for (int j = 0; j < 4; ++j) {
        const int n = n0 + 16 * j + l16;
        #pragma unroll
        for (int v = 0; v < 8; ++v) {
            C[(size_t)(m0 + v + 8 * half) * DIM + n] = acc[j][v];
        }
    }
}

// ---------------- 4) attention logits -> exp weights ------------------------
// qw[r] = exp(dot(Q[r,:], w_att) * inv_sqrt_d + (1-mask[r])*-10000)
__global__ __launch_bounds__(256)
void qw_kernel(const float* __restrict__ Q, const float* __restrict__ w_att,
               const float* __restrict__ mask, float* __restrict__ qw) {
    __shared__ float smem[8];
    const size_t r = blockIdx.x;
    const float* row = Q + r * (size_t)DIM;
    float p = 0.0f;
    for (int i = threadIdx.x; i < DIM; i += blockDim.x) p += row[i] * w_att[i];
    float tot = block_reduce_sum(p, smem);
    if (threadIdx.x == 0) {
        float mterm = (1.0f - mask[r]) * -10000.0f;
        qw[r] = expf(tot * INV_SQRT_D + mterm);
    }
}

// kw[r] = exp(dot(P[r,:]*K[r,:], w_att) * inv_sqrt_d + mask_term)
__global__ __launch_bounds__(256)
void kw_kernel(const float* __restrict__ P, const float* __restrict__ K,
               const float* __restrict__ w_att, const float* __restrict__ mask,
               float* __restrict__ kw) {
    __shared__ float smem[8];
    const size_t r = blockIdx.x;
    const float* pr = P + r * (size_t)DIM;
    const float* kr = K + r * (size_t)DIM;
    float p = 0.0f;
    for (int i = threadIdx.x; i < DIM; i += blockDim.x)
        p += pr[i] * kr[i] * w_att[i];
    float tot = block_reduce_sum(p, smem);
    if (threadIdx.x == 0) {
        float mterm = (1.0f - mask[r]) * -10000.0f;
        kw[r] = expf(tot * INV_SQRT_D + mterm);
    }
}

// ---------------- 5) denominator cumsum -> reciprocal (one wave / batch) ----
// Stores rden[s] = 1 / cumsum(w)[s] so the big scans multiply instead of divide.
__global__ void scan_den_kernel(const float* __restrict__ w,
                                float* __restrict__ rden) {
    const int b = blockIdx.x;
    const float* src = w + (size_t)b * SEQ;
    float* dst = rden + (size_t)b * SEQ;
    const int lane = threadIdx.x;   // 0..31
    float carry = 0.0f;
    for (int s0 = 0; s0 < SEQ; s0 += 32) {
        float v = src[s0 + lane];
        #pragma unroll
        for (int off = 1; off < 32; off <<= 1) {
            float n = __shfl_up(v, off, 32);
            if (lane >= off) v += n;
        }
        v += carry;
        dst[s0 + lane] = 1.0f / v;
        carry = __shfl(v, 31, 32);
    }
}

// ---------------- 6) per-channel pooled scan: P = cumsum(w*X)*rden ----------
__global__ __launch_bounds__(256)
void scan_pool_kernel(const float* __restrict__ w, const float* __restrict__ rden,
                      const float* __restrict__ X, float* __restrict__ P) {
    const int gid = blockIdx.x * blockDim.x + threadIdx.x;  // 0..B*D
    const int b = gid >> 10;          // /DIM
    const int d = gid & (DIM - 1);
    const size_t base = (size_t)b * SEQ * DIM + d;
    const float* wb = w + (size_t)b * SEQ;
    const float* db = rden + (size_t)b * SEQ;
    float acc = 0.0f;
    #pragma unroll 4
    for (int s = 0; s < SEQ; ++s) {
        const size_t idx = base + (size_t)s * DIM;
        acc += wb[s] * X[idx];
        P[idx] = acc * db[s];
    }
}

// ---------------- 7) final scan + output: out = pooled_keys * values --------
__global__ __launch_bounds__(256)
void final_scan_kernel(const float* __restrict__ kw, const float* __restrict__ rden,
                       const float* __restrict__ P, const float* __restrict__ K,
                       const float* __restrict__ Q, float* __restrict__ out) {
    const int gid = blockIdx.x * blockDim.x + threadIdx.x;
    const int b = gid >> 10;
    const int d = gid & (DIM - 1);
    const size_t base = (size_t)b * SEQ * DIM + d;
    const float* wb = kw + (size_t)b * SEQ;
    const float* db = rden + (size_t)b * SEQ;
    float acc = 0.0f;
    #pragma unroll 4
    for (int s = 0; s < SEQ; ++s) {
        const size_t idx = base + (size_t)s * DIM;
        const float mixed = P[idx] * K[idx];       // pooled_query * keys
        acc += wb[s] * mixed;
        out[idx] = (acc * db[s]) * Q[idx];         // pooled_keys * values
    }
}

// ---------------------------------------------------------------------------
extern "C" void kernel_launch(void* const* d_in, const int* in_sizes, int n_in,
                              void* d_out, int out_size, void* d_ws, size_t ws_size,
                              hipStream_t stream) {
    const float* hidden = (const float*)d_in[0];
    const float* amask  = (const float*)d_in[1];
    const float* Wq     = (const float*)d_in[2];
    const float* wq_att = (const float*)d_in[3];
    const float* Wk     = (const float*)d_in[4];
    const float* wk_att = (const float*)d_in[5];
    const float* ln_g   = (const float*)d_in[6];
    const float* ln_b   = (const float*)d_in[7];
    float* out = (float*)d_out;

    // ---- workspace carve-up (256B aligned) ----
    char* w = (char*)d_ws;
    auto alloc = [&](size_t bytes) -> void* {
        void* p = (void*)w;
        w += (bytes + 255) & ~(size_t)255;
        return p;
    };
    __bf16* hbf  = (__bf16*)alloc((size_t)ROWS * DIM * 2);   // 32 MB
    __bf16* wqb  = (__bf16*)alloc((size_t)DIM * DIM * 2);    //  2 MB
    __bf16* wkb  = (__bf16*)alloc((size_t)DIM * DIM * 2);    //  2 MB
    float*  Qbuf = (float*) alloc((size_t)ROWS * DIM * 4);   // 64 MB
    float*  Kbuf = (float*) alloc((size_t)ROWS * DIM * 4);   // 64 MB
    float*  Pbuf = (float*) alloc((size_t)ROWS * DIM * 4);   // 64 MB
    float*  qw   = (float*) alloc((size_t)ROWS * 4);
    float*  kw   = (float*) alloc((size_t)ROWS * 4);
    float*  denQ = (float*) alloc((size_t)ROWS * 4);         // reciprocal cumsum
    float*  denK = (float*) alloc((size_t)ROWS * 4);         // reciprocal cumsum

    // 1) LayerNorm + bf16 cast
    ln_cast_kernel<<<ROWS, 256, 0, stream>>>(hidden, ln_g, ln_b, hbf);
    // 2) weight casts
    castw_kernel<<<(DIM * DIM + 255) / 256, 256, 0, stream>>>(Wq, Wk, wqb, wkb,
                                                              DIM * DIM);
    // 3) Q = h@Wq^T, K = h@Wk^T  (WMMA bf16 -> f32)
    dim3 ggrid(ROWS / 128, DIM / 64, 2);
    gemm_wmma_kernel<<<ggrid, 256, 0, stream>>>(hbf, wqb, wkb, Qbuf, Kbuf);
    // 4) qw weights
    qw_kernel<<<ROWS, 256, 0, stream>>>(Qbuf, wq_att, amask, qw);
    // 5) reciprocal denominator cumsum (qw)
    scan_den_kernel<<<BATCH, 32, 0, stream>>>(qw, denQ);
    // 6) pooled_query scan
    scan_pool_kernel<<<(BATCH * DIM) / 256, 256, 0, stream>>>(qw, denQ, Qbuf, Pbuf);
    // 7) kw weights from pooled_query * keys
    kw_kernel<<<ROWS, 256, 0, stream>>>(Pbuf, Kbuf, wk_att, amask, kw);
    // 8) reciprocal denominator cumsum (kw)
    scan_den_kernel<<<BATCH, 32, 0, stream>>>(kw, denK);
    // 9) final scan + output (pooled_keys * values)
    final_scan_kernel<<<(BATCH * DIM) / 256, 256, 0, stream>>>(kw, denK, Pbuf,
                                                               Kbuf, Qbuf, out);
}
